// KANLinear_65755949302262
// MI455X (gfx1250) — compile-verified, matrix operand order checked by reading, not run on hardware
//
#include <hip/hip_runtime.h>
#include <hip/hip_bf16.h>
#include <math.h>

typedef __attribute__((ext_vector_type(16))) _Float16 v16h;
typedef __attribute__((ext_vector_type(8)))  _Float16 v8h;
typedef __attribute__((ext_vector_type(8)))  float    v8f;

#define IN_F    512
#define OUT_F   512
#define KB      8                   // basis functions per input feature
#define KSPLINE (IN_F * KB)         // 4096
#define KTOT    (KSPLINE + IN_F)    // 4608 = spline K + base K
#define NTOK    8192                // B*S = 8*1024

// ---------------------------------------------------------------------------
// Prep: fused f16 weight matrix  W[o][k]  (o in [0,512), k in [0,4608))
//   k <  4096 : scale_spline[o][k>>3] * coeff[o][k>>3][k&7]  (coeff is contiguous in (i,k))
//   k >= 4096 : scale_base[o][k-4096]
// ---------------------------------------------------------------------------
__global__ void kan_prep_w(const float* __restrict__ coeff,
                           const float* __restrict__ scale_base,
                           const float* __restrict__ scale_spline,
                           _Float16* __restrict__ Wf16) {
  int idx = blockIdx.x * blockDim.x + threadIdx.x;
  if (idx >= OUT_F * KTOT) return;
  int o = idx / KTOT;
  int k = idx - o * KTOT;
  float w;
  if (k < KSPLINE) {
    w = scale_spline[o * IN_F + (k >> 3)] * coeff[(size_t)o * KSPLINE + k];
  } else {
    w = scale_base[o * IN_F + (k - KSPLINE)];
  }
  Wf16[idx] = (_Float16)w;
}

__global__ void kan_prep_bias(const float* __restrict__ base_bias,
                              float* __restrict__ bias) {
  int o = blockIdx.x * blockDim.x + threadIdx.x;
  if (o >= OUT_F) return;
  float s = 0.f;
  for (int i = 0; i < IN_F; ++i) s += base_bias[(size_t)o * IN_F + i];
  bias[o] = s;
}

// ---------------------------------------------------------------------------
// Main fused GEMM:  out[n][o] = sum_k A[n][k] * W[o][k] + bias[o]
//   A[n][k] built on the fly from x:
//     k <  4096 : T_{k&7}(clip(x[n][k>>3]))  (T0=1, T1=x, Tj=2x*T(j-1)-1)
//     k >= 4096 : silu(clip(x[n][k-4096]))
// Block: 256 threads (8 waves), tile 128(M) x 128(N), KC = 64.
// Wave grid 4(M) x 2(N); each wave: 2x4 WMMA 16x16 tiles -> 16 wmma / chunk.
// ---------------------------------------------------------------------------
__global__ __launch_bounds__(256)
void kan_wmma(const float* __restrict__ x,
              const _Float16* __restrict__ W,
              const float* __restrict__ bias,
              float* __restrict__ out) {
  __shared__ __align__(16) _Float16 As[128 * 64];   // 16 KB
  __shared__ __align__(16) _Float16 Bs[128 * 64];   // 16 KB

  const int tid   = threadIdx.x;
  const int lane  = tid & 31;
  const int wave  = tid >> 5;     // 0..7
  const int wm    = wave & 3;     // wave-row: 32 rows each
  const int wn    = wave >> 2;    // wave-col: 64 cols each
  const int row0  = blockIdx.y * 128;
  const int col0  = blockIdx.x * 128;

  const int lhalf = lane >> 4;    // 0 or 1
  const int l15   = lane & 15;
  const int kfrag = lhalf << 3;   // lanes 0-15 take K 0-7/16-23, lanes 16-31 take K 8-15/24-31

  v8f acc[2][4];
  {
    v8f z = {};
#pragma unroll
    for (int i = 0; i < 2; ++i)
#pragma unroll
      for (int j = 0; j < 4; ++j) acc[i][j] = z;
  }

  for (int k0 = 0; k0 < KTOT; k0 += 64) {
    // ---------------- stage A tile (compute basis / silu into LDS) ---------
    if (k0 < KSPLINE) {
      const int fbase = k0 >> 3;   // first of 8 input features in this chunk
#pragma unroll
      for (int it = 0; it < 4; ++it) {
        int p  = it * 256 + tid;   // 0..1023 = 128 rows x 8 features
        int nl = p >> 3;
        int il = p & 7;
        float xv = x[(size_t)(row0 + nl) * IN_F + fbase + il];
        float xc = fminf(1.f, fmaxf(-1.f, xv));
        float t[8];
        t[0] = 1.f; t[1] = xc;
#pragma unroll
        for (int j = 2; j < 8; ++j) t[j] = 2.f * xc * t[j - 1] - 1.f;
        v8h st;
#pragma unroll
        for (int j = 0; j < 8; ++j) st[j] = (_Float16)t[j];
        *(v8h*)&As[nl * 64 + il * 8] = st;   // one ds_store_b128
      }
    } else {
      const int fbase = k0 - KSPLINE;
#pragma unroll 4
      for (int it = 0; it < 32; ++it) {
        int e  = it * 256 + tid;   // 0..8191 = 128 rows x 64 features
        int nl = e >> 6;
        int c  = e & 63;
        float xv = x[(size_t)(row0 + nl) * IN_F + fbase + c];
        float xc = fminf(1.f, fmaxf(-1.f, xv));
        As[nl * 64 + c] = (_Float16)(xc / (1.f + __expf(-xc)));
      }
    }

    // ---------------- stage B tile (f16 weights) ---------------------------
#pragma unroll
    for (int it = 0; it < 4; ++it) {
      int g  = it * 256 + tid;     // 0..1023 groups of 8 halfs
      int ol = g >> 3;             // 0..127 output column
      int kg = (g & 7) * 8;
      v8h wv = *(const v8h*)&W[(size_t)(col0 + ol) * KTOT + k0 + kg];
      *(v8h*)&Bs[ol * 64 + kg] = wv;
    }

    // gfx1250 prefetch of next weight chunk (global_prefetch_b8)
    if (k0 + 64 < KTOT) {
      __builtin_prefetch(&W[(size_t)(col0 + (tid >> 1)) * KTOT + (k0 + 64) + (tid & 1) * 32], 0, 0);
    }

    __syncthreads();

    // ---------------- consume: 2 k-steps of v_wmma_f32_16x16x32_f16 --------
#pragma unroll
    for (int ks = 0; ks < 64; ks += 32) {
      v16h a[2];
#pragma unroll
      for (int ms = 0; ms < 2; ++ms) {
        int r = wm * 32 + ms * 16 + l15;
        v8h lo = *(const v8h*)&As[r * 64 + ks + kfrag];
        v8h hi = *(const v8h*)&As[r * 64 + ks + kfrag + 16];
#pragma unroll
        for (int t = 0; t < 8; ++t) { a[ms][t] = lo[t]; a[ms][8 + t] = hi[t]; }
      }
      v16h b[4];
#pragma unroll
      for (int ns = 0; ns < 4; ++ns) {
        int c = wn * 64 + ns * 16 + l15;
        v8h lo = *(const v8h*)&Bs[c * 64 + ks + kfrag];
        v8h hi = *(const v8h*)&Bs[c * 64 + ks + kfrag + 16];
#pragma unroll
        for (int t = 0; t < 8; ++t) { b[ns][t] = lo[t]; b[ns][8 + t] = hi[t]; }
      }
#pragma unroll
      for (int ms = 0; ms < 2; ++ms)
#pragma unroll
        for (int ns = 0; ns < 4; ++ns)
          acc[ms][ns] = __builtin_amdgcn_wmma_f32_16x16x32_f16(
              false, a[ms], false, b[ns], (short)0, acc[ms][ns], false, false);
    }

    __syncthreads();
  }

  // ---------------- epilogue: + bias, store f32 ----------------------------
  const int mrow = lhalf * 8;   // C layout: VGPR r, lanes16-31 -> M = r + 8
#pragma unroll
  for (int ms = 0; ms < 2; ++ms) {
    int gm = row0 + wm * 32 + ms * 16 + mrow;
#pragma unroll
    for (int ns = 0; ns < 4; ++ns) {
      int gn = col0 + wn * 64 + ns * 16 + l15;
      float bb = bias[gn];
#pragma unroll
      for (int r = 0; r < 8; ++r) {
        out[(size_t)(gm + r) * OUT_F + gn] = acc[ms][ns][r] + bb;
      }
    }
  }
}

// ---------------------------------------------------------------------------
extern "C" void kernel_launch(void* const* d_in, const int* in_sizes, int n_in,
                              void* d_out, int out_size, void* d_ws, size_t ws_size,
                              hipStream_t stream) {
  // setup_inputs() order: x, coeff, scale_base, scale_spline, base_bias (all f32)
  const float* x            = (const float*)d_in[0];
  const float* coeff        = (const float*)d_in[1];
  const float* scale_base   = (const float*)d_in[2];
  const float* scale_spline = (const float*)d_in[3];
  const float* base_bias    = (const float*)d_in[4];
  float* out = (float*)d_out;

  // Workspace layout: f16 fused weights (512 x 4608), then f32 bias (512)
  _Float16* Wf16 = (_Float16*)d_ws;
  float* bias    = (float*)((char*)d_ws + (size_t)OUT_F * KTOT * sizeof(_Float16));

  {
    int total = OUT_F * KTOT;                       // 2,359,296
    kan_prep_w<<<(total + 255) / 256, 256, 0, stream>>>(coeff, scale_base, scale_spline, Wf16);
  }
  kan_prep_bias<<<(OUT_F + 255) / 256, 256, 0, stream>>>(base_bias, bias);

  dim3 grid(OUT_F / 128, NTOK / 128);               // (4, 64)
  kan_wmma<<<grid, 256, 0, stream>>>(x, Wf16, bias, out);
}
